// LSTMCell_19988777796262
// MI455X (gfx1250) — matrix operand order repaired and checked by reading, not compiled, not run on hardware
//
#include <hip/hip_runtime.h>
#include <hip/hip_bf16.h>

// ---------------------------------------------------------------------------
// Types for CDNA5 WMMA
// ---------------------------------------------------------------------------
typedef _Float16 v16h __attribute__((ext_vector_type(16)));
typedef float    v8f  __attribute__((ext_vector_type(8)));
typedef int      v4i  __attribute__((ext_vector_type(4)));

union FragU {
    uint4 q[2];   // two 16-byte chunks
    v16h  v;      // 16 f16 = 32 bytes
};

#define Bsz   4096
#define Hdim  1024
#define Kdim  3072   // I + 2H = 1024 + 2048
#define N7H   7168   // 7*H
#define N2H   2048   // 2*H

#if defined(__has_builtin)
#if __has_builtin(__builtin_amdgcn_global_load_async_to_lds_b128)
#define HAVE_ASYNC_LDS 1
#endif
#endif

#ifdef HAVE_ASYNC_LDS
// builtin signature (from hipcc diagnostics):
//   (v4i __device__(AS1)* src_global, v4i __shared__(AS3)* dst_lds,
//    imm int offset, imm int cpol)
typedef __attribute__((address_space(1))) v4i* gas_ptr;
typedef __attribute__((address_space(3))) v4i* las_ptr;
#endif

// ---------------------------------------------------------------------------
// Pack / convert fp32 -> f16 into a concatenated row-major [rows x ldDst] buf
// ---------------------------------------------------------------------------
__global__ void pack_f16(const float* __restrict__ src, _Float16* __restrict__ dst,
                         int rows, int cols, int ldDst, int colOff) {
    int idx = blockIdx.x * blockDim.x + threadIdx.x;
    if (idx >= rows * cols) return;
    int r = idx / cols;
    int c = idx - r * cols;
    dst[(size_t)r * ldDst + colOff + c] = (_Float16)src[idx];
}

// ---------------------------------------------------------------------------
// C[M,N] = A[M,K] @ W[N,K]^T + bias1[N] + bias2[N]
// A, W f16 row-major (ld = K); C f32 row-major (ld = N).
// Block tile 128x128, BK = 32, 8 waves (wave grid 4M x 2N), each wave
// 2x4 tiles of v_wmma_f32_16x16x32_f16. Double-buffered LDS, fills done
// with GLOBAL_LOAD_ASYNC_TO_LDS_B128 (ASYNCcnt) when available.
// ---------------------------------------------------------------------------
#define LDT 40   // padded LDS row stride in halves (32 + 8) -> conflict-free

__global__ __launch_bounds__(256) void gemm_f16_wmma(
    const _Float16* __restrict__ A, const _Float16* __restrict__ W,
    const float* __restrict__ bias1, const float* __restrict__ bias2,
    float* __restrict__ C, int M, int N, int K)
{
    __shared__ __align__(16) _Float16 sA[2][128 * LDT];
    __shared__ __align__(16) _Float16 sW[2][128 * LDT];

    const int tid  = threadIdx.x;
    const int wave = tid >> 5;
    const int lane = tid & 31;
    const int wm   = wave & 3;   // 0..3  -> 32-row strip
    const int wn   = wave >> 2;  // 0..1  -> 64-col strip
    const int blockM = blockIdx.y * 128;
    const int blockN = blockIdx.x * 128;

    v8f acc[2][4] = {};

    const int lrow = lane & 15;
    const int kcA  = (lane >> 4) * 8;    // A frag K chunk base (0 or 8)
    const int kcB  = (lane >> 4) * 8;    // B frag K chunk base (0 or 8)

    // this thread's two 16B chunks of each 128x32 tile
    const int row0 = tid >> 2;                   // chunk tid       (0..255)
    const int col0 = (tid & 3) * 8;
    const int row1 = (tid + 256) >> 2;           // chunk tid + 256
    const int col1 = ((tid + 256) & 3) * 8;

    // stage one 128x32 tile of A and W into LDS buffer `buf` at K offset k0
    auto stage = [&](int buf, int k0) {
#ifdef HAVE_ASYNC_LDS
        __builtin_amdgcn_global_load_async_to_lds_b128(
            (gas_ptr)&A[(size_t)(blockM + row0) * K + k0 + col0],
            (las_ptr)&sA[buf][row0 * LDT + col0], 0, 0);
        __builtin_amdgcn_global_load_async_to_lds_b128(
            (gas_ptr)&A[(size_t)(blockM + row1) * K + k0 + col1],
            (las_ptr)&sA[buf][row1 * LDT + col1], 0, 0);
        __builtin_amdgcn_global_load_async_to_lds_b128(
            (gas_ptr)&W[(size_t)(blockN + row0) * K + k0 + col0],
            (las_ptr)&sW[buf][row0 * LDT + col0], 0, 0);
        __builtin_amdgcn_global_load_async_to_lds_b128(
            (gas_ptr)&W[(size_t)(blockN + row1) * K + k0 + col1],
            (las_ptr)&sW[buf][row1 * LDT + col1], 0, 0);
#else
        *reinterpret_cast<uint4*>(&sA[buf][row0 * LDT + col0]) =
            *reinterpret_cast<const uint4*>(&A[(size_t)(blockM + row0) * K + k0 + col0]);
        *reinterpret_cast<uint4*>(&sA[buf][row1 * LDT + col1]) =
            *reinterpret_cast<const uint4*>(&A[(size_t)(blockM + row1) * K + k0 + col1]);
        *reinterpret_cast<uint4*>(&sW[buf][row0 * LDT + col0]) =
            *reinterpret_cast<const uint4*>(&W[(size_t)(blockN + row0) * K + k0 + col0]);
        *reinterpret_cast<uint4*>(&sW[buf][row1 * LDT + col1]) =
            *reinterpret_cast<const uint4*>(&W[(size_t)(blockN + row1) * K + k0 + col1]);
#endif
    };

    const int KT = K >> 5;   // 96 K-steps of 32
    stage(0, 0);

    for (int kt = 0; kt < KT; ++kt) {
        const int cur = kt & 1;
        if (kt + 1 < KT) {
            stage(cur ^ 1, (kt + 1) * 32);   // prefetch next tile (other buffer)
#ifdef HAVE_ASYNC_LDS
            // oldest 4 async ops (current buffer) must be done; newest 4 may fly
            asm volatile("s_wait_asynccnt 0x4" ::: "memory");
#endif
        } else {
#ifdef HAVE_ASYNC_LDS
            asm volatile("s_wait_asynccnt 0x0" ::: "memory");
#endif
        }
        __syncthreads();

        // ---- build fragments from LDS buffer `cur`
        FragU a[2];
#pragma unroll
        for (int i = 0; i < 2; ++i) {
            int r = wm * 32 + i * 16 + lrow;
            a[i].q[0] = *reinterpret_cast<const uint4*>(&sA[cur][r * LDT + kcA]);
            a[i].q[1] = *reinterpret_cast<const uint4*>(&sA[cur][r * LDT + kcA + 16]);
        }
        FragU b[4];
#pragma unroll
        for (int j = 0; j < 4; ++j) {
            int r = wn * 64 + j * 16 + lrow;
            b[j].q[0] = *reinterpret_cast<const uint4*>(&sW[cur][r * LDT + kcB]);
            b[j].q[1] = *reinterpret_cast<const uint4*>(&sW[cur][r * LDT + kcB + 16]);
        }

        // ---- 8 WMMAs: acc[i][j] += a[i] x b[j]
#pragma unroll
        for (int i = 0; i < 2; ++i)
#pragma unroll
            for (int j = 0; j < 4; ++j)
                acc[i][j] = __builtin_amdgcn_wmma_f32_16x16x32_f16(
                    false, a[i].v, false, b[j].v,
                    (short)0, acc[i][j], false, false);

        __syncthreads();   // all reads of buffer `cur` done before it is refilled
    }

    // ---- epilogue: add both biases, write f32
    const int rhalf = (lane >> 4) * 8;   // C layout: lanes>=16 hold M+8
#pragma unroll
    for (int j = 0; j < 4; ++j) {
        int col = blockN + wn * 64 + j * 16 + (lane & 15);
        float bsum = bias1[col] + bias2[col];
#pragma unroll
        for (int i = 0; i < 2; ++i) {
            int rowb = blockM + wm * 32 + i * 16 + rhalf;
#pragma unroll
            for (int v = 0; v < 8; ++v)
                C[(size_t)(rowb + v) * N + col] = acc[i][j][v] + bsum;
        }
    }
}

// ---------------------------------------------------------------------------
// Element-wise tail: gates, complex bound / mul, outputs h_t then c_t
// ---------------------------------------------------------------------------
__device__ __forceinline__ float sigmoidf(float x) {
    return 1.0f / (1.0f + __expf(-x));
}
__device__ __forceinline__ void bound2(float& r, float& i) {
    float d = fmaxf(sqrtf(r * r + i * i), 1.0f);
    float inv = 1.0f / d;
    r *= inv; i *= inv;
}

__global__ __launch_bounds__(256) void lstm_tail(
    const float* __restrict__ pre,   // B x 7H
    const float* __restrict__ upre,  // B x 2H
    const float* __restrict__ c,     // B x 2H
    float* __restrict__ out)         // [h_t (B x 2H) | c_t (B x 2H)]
{
    int idx = blockIdx.x * blockDim.x + threadIdx.x;   // B*H threads
    int b = idx >> 10;          // / H
    int j = idx & (Hdim - 1);   // % H
    if (b >= Bsz) return;

    const float* pr = pre + (size_t)b * N7H;
    float gf = sigmoidf(pr[j]);
    float gi = sigmoidf(pr[Hdim + j]);
    float go = sigmoidf(pr[2 * Hdim + j]);
    float ri_r = pr[3 * Hdim + j], ri_i = pr[4 * Hdim + j];
    float ro_r = pr[5 * Hdim + j], ro_i = pr[6 * Hdim + j];

    const float* up = upre + (size_t)b * N2H;
    float u_r = up[j], u_i = up[Hdim + j];

    bound2(u_r, u_i);
    bound2(ri_r, ri_i);
    bound2(ro_r, ro_i);

    float gu_r = gi * u_r, gu_i = gi * u_i;

    const float* cp = c + (size_t)b * N2H;
    float ct_r = gf * cp[j]        + (ri_r * gu_r - ri_i * gu_i);
    float ct_i = gf * cp[Hdim + j] + (ri_r * gu_i + ri_i * gu_r);

    float m_r = ro_r * ct_r - ro_i * ct_i;
    float m_i = ro_r * ct_i + ro_i * ct_r;
    bound2(m_r, m_i);

    size_t ho = (size_t)b * N2H;
    out[ho + j]        = go * m_r;
    out[ho + Hdim + j] = go * m_i;
    size_t co = (size_t)Bsz * N2H + ho;
    out[co + j]        = ct_r;
    out[co + Hdim + j] = ct_i;
}

// ---------------------------------------------------------------------------
// kernel_launch
// ---------------------------------------------------------------------------
extern "C" void kernel_launch(void* const* d_in, const int* in_sizes, int n_in,
                              void* d_out, int out_size, void* d_ws, size_t ws_size,
                              hipStream_t stream) {
    const float* x     = (const float*)d_in[0];
    const float* h     = (const float*)d_in[1];
    const float* c     = (const float*)d_in[2];
    const float* w_i2h = (const float*)d_in[3];
    const float* b_i2h = (const float*)d_in[4];
    const float* w_h2h = (const float*)d_in[5];
    const float* b_h2h = (const float*)d_in[6];
    const float* w_i2u = (const float*)d_in[7];
    const float* b_i2u = (const float*)d_in[8];
    const float* w_h2u = (const float*)d_in[9];
    const float* b_h2u = (const float*)d_in[10];
    float* out = (float*)d_out;

    char* ws = (char*)d_ws;
    size_t off = 0;
    auto carve = [&](size_t bytes) -> void* {
        void* p = ws + off;
        off = (off + bytes + 255) & ~(size_t)255;
        return p;
    };
    _Float16* Apk = (_Float16*)carve((size_t)Bsz * Kdim * 2);   // [x|h]      f16
    _Float16* Whp = (_Float16*)carve((size_t)N7H * Kdim * 2);   // [i2h|h2h]  f16
    _Float16* Wup = (_Float16*)carve((size_t)N2H * Kdim * 2);   // [i2u|h2u]  f16
    float*    pre = (float*)carve((size_t)Bsz * N7H * 4);       // preact f32
    float*    upr = (float*)carve((size_t)Bsz * N2H * 4);       // u_pre  f32

    auto blocks = [](long n) { return (int)((n + 255) / 256); };

    // pack: activations and weights, K-concatenated
    pack_f16<<<blocks((long)Bsz * 1024), 256, 0, stream>>>(x,     Apk, Bsz, 1024, Kdim, 0);
    pack_f16<<<blocks((long)Bsz * 2048), 256, 0, stream>>>(h,     Apk, Bsz, 2048, Kdim, 1024);
    pack_f16<<<blocks((long)N7H * 1024), 256, 0, stream>>>(w_i2h, Whp, N7H, 1024, Kdim, 0);
    pack_f16<<<blocks((long)N7H * 2048), 256, 0, stream>>>(w_h2h, Whp, N7H, 2048, Kdim, 1024);
    pack_f16<<<blocks((long)N2H * 1024), 256, 0, stream>>>(w_i2u, Wup, N2H, 1024, Kdim, 0);
    pack_f16<<<blocks((long)N2H * 2048), 256, 0, stream>>>(w_h2u, Wup, N2H, 2048, Kdim, 1024);

    // GEMMs (WMMA): preact = Apk @ Whp^T + biases ; u_pre = Apk @ Wup^T + biases
    gemm_f16_wmma<<<dim3(N7H / 128, Bsz / 128), 256, 0, stream>>>(
        Apk, Whp, b_i2h, b_h2h, pre, Bsz, N7H, Kdim);
    gemm_f16_wmma<<<dim3(N2H / 128, Bsz / 128), 256, 0, stream>>>(
        Apk, Wup, b_i2u, b_h2u, upr, Bsz, N2H, Kdim);

    // fused element-wise tail -> out = [h_t | c_t]
    lstm_tail<<<blocks((long)Bsz * Hdim), 256, 0, stream>>>(pre, upr, c, out);
}
